// MLSTMBlock_87686052315890
// MI455X (gfx1250) — matrix operand order repaired and checked by reading, not compiled, not run on hardware
//
#include <hip/hip_runtime.h>
#include <hip/hip_bf16.h>

typedef __attribute__((ext_vector_type(16))) __bf16 bf16x16;
typedef __attribute__((ext_vector_type(8)))  float  f32x8;
typedef __attribute__((ext_vector_type(4)))  unsigned int u32x4;
typedef __attribute__((ext_vector_type(8)))  int i32x8;
typedef __attribute__((ext_vector_type(4)))  int i32x4;

#define T_  1024
#define DM_ 128
#define DI_ 256
#define D_  256
#define H_  4
#define HD_ 64
#define L_  32      // chunk length (matches WMMA K=32)
#define NC_ 32      // number of chunks

#if defined(__gfx1250__) && __has_builtin(__builtin_amdgcn_tensor_load_to_lds)
#define USE_TDM 1
#else
#define USE_TDM 0
#endif

// ---------------- WMMA helpers (CDNA5 16x16x32 bf16, fp32 accum) ----------------

__device__ __forceinline__ int kidxA(int e, int kh){
  // A 16x32 k order chosen here (B matches; not AMD's table order): lanes 0-15 rows M, kh=lane>>4 selects contiguous K-half (k = 16*kh + e)
  return kh*16 + e;
}

__device__ __forceinline__ f32x8 wmma_bf16(bf16x16 a, bf16x16 b, f32x8 c){
  return __builtin_amdgcn_wmma_f32_16x16x32_bf16(false, a, false, b, (short)0, c, false, false);
}

// A fragment 16x32 from row-major bf16 src (A[m][k] = p[m*ld+k])
__device__ __forceinline__ bf16x16 ldA(const __bf16* p, int ld, int row0, int k0, int lane){
  int m = lane & 15, kh = lane >> 4;
  bf16x16 a;
#pragma unroll
  for (int e = 0; e < 16; ++e) a[e] = p[(row0 + m)*ld + k0 + kidxA(e, kh)];
  return a;
}
// A fragment where logical A[m][k] = src[k*ld + m] (transposed source)
__device__ __forceinline__ bf16x16 ldA_T(const __bf16* p, int ld, int row0, int k0, int lane){
  int m = lane & 15, kh = lane >> 4;
  bf16x16 a;
#pragma unroll
  for (int e = 0; e < 16; ++e) a[e] = p[(k0 + kidxA(e, kh))*ld + row0 + m];
  return a;
}
// B fragment 32x16 from row-major bf16 src (B[k][n] = p[k*ld+n])
__device__ __forceinline__ bf16x16 ldB(const __bf16* p, int ld, int k0, int n0, int lane){
  int n = lane & 15, kh = lane >> 4;
  bf16x16 b;
#pragma unroll
  for (int e = 0; e < 16; ++e) b[e] = p[(k0 + kh*16 + e)*ld + n0 + n];
  return b;
}
// B fragment where logical B[k][n] = src[n*ld + k]
__device__ __forceinline__ bf16x16 ldB_T(const __bf16* p, int ld, int k0, int n0, int lane){
  int n = lane & 15, kh = lane >> 4;
  bf16x16 b;
#pragma unroll
  for (int e = 0; e < 16; ++e) b[e] = p[(n0 + n)*ld + k0 + kh*16 + e];
  return b;
}

#if USE_TDM
// Issue a 2D Tensor-Data-Mover load: rows x cols bf16 tile (row-major, stride in elems)
// from global into LDS at byte offset lds_off.  Per CDNA5 ISA D# groups 0/1.
// clang-23 toolchain: 6-arg builtin (g0, g1, g2, g3, g4, cpol).
__device__ __forceinline__ void tdm_load_2d_bf16(unsigned lds_off, const void* gptr,
                                                 int rows, int cols, int stride_elems){
  unsigned long long ga = (unsigned long long)(size_t)gptr;
  u32x4 g0;
  g0[0] = 1u;                                               // count=1, user mode
  g0[1] = lds_off;                                          // lds_addr (bytes)
  g0[2] = (unsigned)(ga & 0xffffffffu);                     // global_addr[31:0]
  g0[3] = (unsigned)((ga >> 32) & 0x01ffffffu) | (2u << 30);// global_addr[56:32], type=2
  i32x8 g1;
  g1[0] = (1 << 16);                                        // data_size=1 (2 bytes/elem)
  g1[1] = (cols & 0xffff) << 16;                            // tensor_dim0[15:0]
  g1[2] = ((cols >> 16) & 0xffff) | ((rows & 0xffff) << 16);// tensor_dim0 hi / tensor_dim1 lo
  g1[3] = ((rows >> 16) & 0xffff) | ((cols & 0xffff) << 16);// tensor_dim1 hi / tile_dim0
  g1[4] = rows & 0xffff;                                    // tile_dim1 (tile_dim2=0)
  g1[5] = stride_elems;                                     // tensor_dim0_stride[31:0]
  g1[6] = 0;
  g1[7] = 0;
  i32x4 z4 = {0, 0, 0, 0};
  i32x8 z8 = {0, 0, 0, 0, 0, 0, 0, 0};
  __builtin_amdgcn_tensor_load_to_lds(g0, g1, z4, z4, z8, 0);
}
#endif

// ---------------- small kernels ----------------

__global__ __launch_bounds__(256) void cvt_f32_bf16(const float* __restrict__ s, __bf16* __restrict__ d, int n){
  int i = blockIdx.x*256 + threadIdx.x;
  if (i < n) d[i] = (__bf16)s[i];
}

__global__ __launch_bounds__(128) void ln_kernel(const float* __restrict__ x,
                                                 const float* __restrict__ sc,
                                                 const float* __restrict__ bi,
                                                 __bf16* __restrict__ xn){
  __shared__ float red[128];
  int t = blockIdx.x, c = threadIdx.x;
  float v = x[t*DM_ + c];
  red[c] = v; __syncthreads();
  for (int s = 64; s > 0; s >>= 1){ if (c < s) red[c] += red[c+s]; __syncthreads(); }
  float mu = red[0] / (float)DM_; __syncthreads();
  float dv = v - mu;
  red[c] = dv*dv; __syncthreads();
  for (int s = 64; s > 0; s >>= 1){ if (c < s) red[c] += red[c+s]; __syncthreads(); }
  float var = red[0] / (float)DM_;
  float r = rsqrtf(var + 1e-6f);
  xn[t*DM_ + c] = (__bf16)(dv*r*sc[c] + bi[c]);
}

// ---------------- generic WMMA GEMM: out = epi((A@B + bias)*scale) ----------------
// act: 0 none, 1 silu, 2 sigmoid, 3 exp.  bd: block-diagonal (per-head 64x64) mode.
// tnshift = log2(N/16); bmask = bias length - 1 (power of two).
__global__ __launch_bounds__(256) void gemm_bf16_kernel(
    const __bf16* __restrict__ A, int lda,
    const __bf16* __restrict__ Bm, int ldb,
    const float* __restrict__ bias, int bmask,
    float scale, int act,
    const float* __restrict__ mul, const float* __restrict__ addp,
    float* __restrict__ outF, __bf16* __restrict__ outBf,
    int M, int N, int K, int tnshift, int bd){
  int tile = blockIdx.x*8 + threadIdx.y;
  if (tile >= (M >> 4) << tnshift) return;
  int tm = tile >> tnshift, tn = tile & ((1 << tnshift) - 1);
  int lane = threadIdx.x, n = lane & 15, kh = lane >> 4;
  int acb = bd ? ((tn << 4) & ~63) : 0;   // A column base for block-diagonal mode
  f32x8 acc = {0.f,0.f,0.f,0.f,0.f,0.f,0.f,0.f};
  for (int k0 = 0; k0 < K; k0 += 32){
    if (k0 + 32 < K) __builtin_prefetch(&A[((tm<<4) + n)*lda + acb + k0 + 32], 0, 3);
    bf16x16 a = ldA(A, lda, tm << 4, acb + k0, lane);
    bf16x16 b = ldB(Bm, ldb, k0, tn << 4, lane);
    acc = wmma_bf16(a, b, acc);
  }
#pragma unroll
  for (int r = 0; r < 8; ++r){
    int gm = (tm << 4) + kh*8 + r, gn = (tn << 4) + n;
    float v2 = acc[r];
    if (bias) v2 += bias[gn & bmask];
    v2 *= scale;
    if (act == 1)      v2 = v2 / (1.f + __expf(-v2));
    else if (act == 2) v2 = 1.f / (1.f + __expf(-v2));
    else if (act == 3) v2 = __expf(v2);
    if (mul)  v2 *= mul[gm*N + gn];
    if (addp) v2 += addp[gm*N + gn];
    if (outF)  outF[gm*N + gn] = v2;
    if (outBf) outBf[gm*N + gn] = (__bf16)v2;
  }
}

// ---------------- time-conv (KW=4, SAME, pad_lo=1) as gathered WMMA GEMM ----------------
__global__ __launch_bounds__(256) void conv_kernel(
    const __bf16* __restrict__ xu, const __bf16* __restrict__ Wc,
    const float* __restrict__ cb, float* __restrict__ outF, __bf16* __restrict__ outBf){
  int tile = blockIdx.x*8 + threadIdx.y;  // 64*16 tiles
  int tm = tile >> 4, tn = tile & 15;
  int lane = threadIdx.x, n = lane & 15, kh = lane >> 4, m = lane & 15;
  f32x8 acc = {0.f,0.f,0.f,0.f,0.f,0.f,0.f,0.f};
  for (int w = 0; w < 4; ++w){
    int row = (tm << 4) + m + w - 1;
    bool ok = (row >= 0) && (row < T_);
    const __bf16* Bp = Wc + w*DI_*DI_;
    for (int k0 = 0; k0 < DI_; k0 += 32){
      bf16x16 a;
#pragma unroll
      for (int e = 0; e < 16; ++e) a[e] = ok ? xu[row*DI_ + k0 + kidxA(e, kh)] : (__bf16)0.f;
      bf16x16 b = ldB(Bp, DI_, k0, tn << 4, lane);
      acc = wmma_bf16(a, b, acc);
    }
  }
#pragma unroll
  for (int r = 0; r < 8; ++r){
    int gm = (tm << 4) + kh*8 + r, gn = (tn << 4) + n;
    float v2 = acc[r] + cb[gn];
    v2 = v2 / (1.f + __expf(-v2));            // silu
    outF[gm*DI_ + gn] = v2;
    outBf[gm*DI_ + gn] = (__bf16)v2;
  }
}

// ---------------- S1: per-chunk local state Cl = (V.*aE/a)^T @ K, plus aE, nl ----------------
__global__ __launch_bounds__(256) void s1_kernel(
    const float* __restrict__ vv, const float* __restrict__ kk,
    const float* __restrict__ it, const float* __restrict__ zf,
    float* __restrict__ Cl, float* __restrict__ aE, float* __restrict__ nl){
  __shared__ __bf16 sV[L_*D_];
  __shared__ __bf16 sK[L_*D_];
  int c = blockIdx.x;
  int tid = threadIdx.y*32 + threadIdx.x;
  { int i = tid; float rc = 0.f, nla = 0.f;
    for (int s = L_-1; s >= 0; --s){
      int g = (c*L_ + s)*D_ + i;
      float ratio = __expf(rc);                 // prod_{r>s} f[r][i]
      sV[s*D_ + i] = (__bf16)(vv[g] * ratio);
      sK[s*D_ + i] = (__bf16)kk[g];
      nla += it[g]*kk[g]*ratio;
      rc += zf[g];
    }
    aE[c*D_ + i] = __expf(rc);
    nl[c*D_ + i] = nla;
  }
  __syncthreads();
  int lane = threadIdx.x, n = lane & 15, kh = lane >> 4;
  for (int tile = threadIdx.y; tile < 256; tile += 8){
    int tm = tile >> 4, tn = tile & 15;
    bf16x16 a = ldA_T(sV, D_, tm << 4, 0, lane);     // A[i][s] = Vtil[s][i]
    bf16x16 b = ldB  (sK, D_, 0, tn << 4, lane);
    f32x8 acc = {0.f,0.f,0.f,0.f,0.f,0.f,0.f,0.f};
    acc = wmma_bf16(a, b, acc);
#pragma unroll
    for (int r = 0; r < 8; ++r)
      Cl[(size_t)c*D_*D_ + ((tm<<4)+kh*8+r)*D_ + (tn<<4)+n] = acc[r];
  }
}

// ---------------- S2: sequential carry scan over chunks; emits final (C, n) states ----------------
__global__ __launch_bounds__(256) void s2_kernel(
    const float* __restrict__ Cl, const float* __restrict__ aE, const float* __restrict__ nl,
    __bf16* __restrict__ CTh, float* __restrict__ npre,
    float* __restrict__ outC, float* __restrict__ outN){
  int j = blockIdx.x, i = threadIdx.x;
  float cv = 0.f;
  for (int c = 0; c < NC_; ++c){
    CTh[(size_t)c*D_*D_ + j*D_ + i] = (__bf16)cv;              // C_prev transposed: CT[j][i]
    cv = aE[c*D_ + i]*cv + Cl[(size_t)c*D_*D_ + i*D_ + j];
  }
  outC[i*D_ + j] = cv;                                          // ct_s[:, -1] row-major [i][j]
  if (j == 0){
    float nv = 1.f;
    for (int c = 0; c < NC_; ++c){
      npre[c*D_ + i] = nv;
      nv = aE[c*D_ + i]*nv + nl[c*D_ + i];
    }
    outN[i] = nv;
  }
}

// ---------------- S3: per-chunk outputs ht via WMMA (inter + intra + denominator) ----------------
__global__ __launch_bounds__(256) void s3_kernel(
    const float* __restrict__ q,  const float* __restrict__ kk,
    const float* __restrict__ vv, const float* __restrict__ it,
    const float* __restrict__ zf, const float* __restrict__ ot,
    const __bf16* __restrict__ CTh, const float* __restrict__ npre,
    float* __restrict__ aAll, float* __restrict__ ht){
  __shared__ __bf16 sA[L_*D_];      // K / knb / vb (per phase)
  __shared__ __bf16 sB[L_*D_];      // Q / qa / Q (per phase)
  __shared__ __bf16 sCT[L_*D_];     // TDM-staged C_prev^T K-slab
  __shared__ __bf16 sS[L_*L_];      // masked S (bf16)
  __shared__ float  sD[L_*L_];      // raw qa@knb^T
  __shared__ float  sDen[L_];
  int c = blockIdx.x;
  int tid = threadIdx.y*32 + threadIdx.x;
  int lane = threadIdx.x, n = lane & 15, kh = lane >> 4;
  const __bf16* CThc = CTh + (size_t)c*D_*D_;

  // P0: decay a[t][i] = exp(cumsum zf) -> global scratch; Q, K -> LDS
  { int i = tid; float cum = 0.f;
    for (int t = 0; t < L_; ++t){
      int g = (c*L_ + t)*D_ + i;
      cum += zf[g];
      aAll[g] = __expf(cum);
      sB[t*D_ + i] = (__bf16)q[g];
      sA[t*D_ + i] = (__bf16)kk[g];
    }
  }
  __threadfence_block();
  __syncthreads();
  // P1: S = Q @ K^T, causal-masked (s<=t)
  if (threadIdx.y < 4){
    int tm = threadIdx.y >> 1, tn = threadIdx.y & 1;
    f32x8 acc = {0.f,0.f,0.f,0.f,0.f,0.f,0.f,0.f};
    for (int k0 = 0; k0 < D_; k0 += 32){
      bf16x16 a = ldA  (sB, D_, tm << 4, k0, lane);
      bf16x16 b = ldB_T(sA, D_, k0, tn << 4, lane);
      acc = wmma_bf16(a, b, acc);
    }
#pragma unroll
    for (int r = 0; r < 8; ++r){
      int t = (tm<<4)+kh*8+r, s = (tn<<4)+n;
      sS[t*L_ + s] = (__bf16)((s <= t) ? acc[r] : 0.f);
    }
  }
  __syncthreads();
  // P2: qa = q.*a, knb = it.*k./a ; Dmat = qa @ knb^T ; denominator
  { int i = tid;
    for (int t = 0; t < L_; ++t){
      int g = (c*L_ + t)*D_ + i;
      float av = aAll[g];
      sA[t*D_ + i] = (__bf16)(it[g]*kk[g]/av);
      sB[t*D_ + i] = (__bf16)(q[g]*av);
    }
  }
  __syncthreads();
  if (threadIdx.y < 4){
    int tm = threadIdx.y >> 1, tn = threadIdx.y & 1;
    f32x8 acc = {0.f,0.f,0.f,0.f,0.f,0.f,0.f,0.f};
    for (int k0 = 0; k0 < D_; k0 += 32){
      bf16x16 a = ldA  (sB, D_, tm << 4, k0, lane);
      bf16x16 b = ldB_T(sA, D_, k0, tn << 4, lane);
      acc = wmma_bf16(a, b, acc);
    }
#pragma unroll
    for (int r = 0; r < 8; ++r){
      int t = (tm<<4)+kh*8+r, s = (tn<<4)+n;
      sD[t*L_ + s] = acc[r];
    }
  }
  __syncthreads();
  if (tid < L_){
    int t = tid; float d = 0.f;
    for (int s = 0; s <= t; ++s) d += sD[t*L_ + s];
    for (int j = 0; j < D_; ++j) d += (float)sB[t*D_ + j] * npre[c*D_ + j];
    sDen[t] = fmaxf(fabsf(d), 1.f);
  }
  __syncthreads();
  // P3: vb = v./a ; H = Q @ C_prev^T + S @ vb ; scale by a, /den, *ot
  { int i = tid;
    for (int t = 0; t < L_; ++t){
      int g = (c*L_ + t)*D_ + i;
      sA[t*D_ + i] = (__bf16)(vv[g] / aAll[g]);
      sB[t*D_ + i] = (__bf16)q[g];
    }
  }
  __syncthreads();
  f32x8 acc[4];
#pragma unroll
  for (int tt = 0; tt < 4; ++tt) acc[tt] = (f32x8){0.f,0.f,0.f,0.f,0.f,0.f,0.f,0.f};
  for (int k0 = 0; k0 < D_; k0 += 32){
#if USE_TDM
    if (threadIdx.y == 0){
      tdm_load_2d_bf16((unsigned)(size_t)&sCT[0], CThc + (size_t)k0*D_, L_, D_, D_);
      __builtin_amdgcn_s_wait_tensorcnt(0);
    }
    __syncthreads();
    const __bf16* Bsrc = sCT; int kb = 0;
#else
    const __bf16* Bsrc = CThc; int kb = k0;
#endif
#pragma unroll
    for (int tt = 0; tt < 4; ++tt){
      int tile = threadIdx.y + (tt << 3);
      int tm = tile >> 4, tn = tile & 15;
      bf16x16 a = ldA(sB, D_, tm << 4, k0, lane);
      bf16x16 b = ldB(Bsrc, D_, kb, tn << 4, lane);   // B[j][i] = C_prev[i][j]
      acc[tt] = wmma_bf16(a, b, acc[tt]);
    }
#if USE_TDM
    __syncthreads();
#endif
  }
#pragma unroll
  for (int tt = 0; tt < 4; ++tt){
    int tile = threadIdx.y + (tt << 3);
    int tm = tile >> 4, tn = tile & 15;
    bf16x16 a2 = ldA(sS, L_, tm << 4, 0, lane);
    bf16x16 b2 = ldB(sA, D_, 0, tn << 4, lane);
    f32x8 av = wmma_bf16(a2, b2, acc[tt]);
#pragma unroll
    for (int r = 0; r < 8; ++r){
      int t = (tm<<4)+kh*8+r, i2 = (tn<<4)+n;
      int g = (c*L_ + t)*D_ + i2;
      ht[g] = av[r] * aAll[g] / sDen[t] * ot[g];
    }
  }
}

// ---------------- group-norm stats (global over T and HD per head) ----------------
__global__ __launch_bounds__(256) void gn_stats_kernel(const float* __restrict__ ht, float* __restrict__ stats){
  __shared__ float r1[256], r2[256];
  int h = blockIdx.x, tid = threadIdx.x;
  float a = 0.f, b = 0.f;
  for (int idx = tid; idx < T_*HD_; idx += 256){
    int t = idx >> 6, cc = h*HD_ + (idx & 63);
    float v = ht[t*D_ + cc];
    a += v; b += v*v;
  }
  r1[tid] = a; r2[tid] = b; __syncthreads();
  for (int s = 128; s > 0; s >>= 1){ if (tid < s){ r1[tid]+=r1[tid+s]; r2[tid]+=r2[tid+s]; } __syncthreads(); }
  if (tid == 0){
    float mu = r1[0] / (float)(T_*HD_);
    float var = r2[0] / (float)(T_*HD_) - mu*mu;
    stats[h*2] = mu; stats[h*2+1] = rsqrtf(var + 1e-6f);
  }
}

__global__ __launch_bounds__(256) void build_h_kernel(
    const float* __restrict__ ht, const float* __restrict__ qk, const float* __restrict__ skip2,
    const float* __restrict__ gsc, const float* __restrict__ gbi,
    const float* __restrict__ stats, __bf16* __restrict__ hbf){
  int t = blockIdx.x, d = threadIdx.x;
  int h = d >> 6;
  float v = (ht[t*D_ + d] - stats[h*2]) * stats[h*2+1] * gsc[d] + gbi[d];
  v = (v + qk[t*D_ + d]) * skip2[t*D_ + d];
  hbf[t*D_ + d] = (__bf16)v;
}

// ---------------- host launcher ----------------

extern "C" void kernel_launch(void* const* d_in, const int* in_sizes, int n_in,
                              void* d_out, int out_size, void* d_ws, size_t ws_size,
                              hipStream_t stream){
  const float* x       = (const float*)d_in[0];
  const float* ln_s    = (const float*)d_in[1];
  const float* ln_b    = (const float*)d_in[2];
  const float* w_skip2 = (const float*)d_in[3];
  const float* b_skip2 = (const float*)d_in[4];
  const float* w_up    = (const float*)d_in[5];
  const float* b_up    = (const float*)d_in[6];
  const float* conv_k  = (const float*)d_in[7];
  const float* conv_b  = (const float*)d_in[8];
  const float* wi      = (const float*)d_in[9];
  const float* bi      = (const float*)d_in[10];
  const float* wf      = (const float*)d_in[11];
  const float* bff     = (const float*)d_in[12];
  const float* wo      = (const float*)d_in[13];
  const float* bo      = (const float*)d_in[14];
  const float* wq      = (const float*)d_in[15];
  const float* bq      = (const float*)d_in[16];
  const float* wk      = (const float*)d_in[17];
  const float* bk      = (const float*)d_in[18];
  const float* wv      = (const float*)d_in[19];
  const float* bv      = (const float*)d_in[20];
  const float* gn_s    = (const float*)d_in[21];
  const float* gn_b    = (const float*)d_in[22];
  const float* w_down  = (const float*)d_in[23];
  const float* b_down  = (const float*)d_in[24];
  (void)in_sizes; (void)n_in; (void)out_size; (void)ws_size;

  float* outp = (float*)d_out;
  float* outC = outp + T_*DM_;             // ct_s[:, -1]  (256x256)
  float* outN = outC + D_*D_;              // nt_s[:, -1]  (256)

  size_t off = 0;
  char* base = (char*)d_ws;
  auto af = [&](size_t n)->float*  { float*  p = (float*)(base + off); off += ((n*4 + 255)/256)*256; return p; };
  auto ah = [&](size_t n)->__bf16* { __bf16* p = (__bf16*)(base + off); off += ((n*2 + 255)/256)*256; return p; };

  float* skip2 = af((size_t)T_*DI_);
  float* xu    = af((size_t)T_*DI_);
  float* qk    = af((size_t)T_*DI_);
  float* itb   = af((size_t)T_*DI_);
  float* zf    = af((size_t)T_*DI_);
  float* otb   = af((size_t)T_*DI_);
  float* qb    = af((size_t)T_*D_);
  float* kb    = af((size_t)T_*D_);
  float* vb    = af((size_t)T_*D_);
  float* htb   = af((size_t)T_*D_);
  float* aAll  = af((size_t)T_*D_);
  float* Cl    = af((size_t)NC_*D_*D_);
  float* aE    = af((size_t)NC_*D_);
  float* nl    = af((size_t)NC_*D_);
  float* npre  = af((size_t)NC_*D_);
  float* gst   = af(8);

  __bf16* CTh     = ah((size_t)NC_*D_*D_);
  __bf16* xn_h    = ah((size_t)T_*DM_);
  __bf16* xu_h    = ah((size_t)T_*DI_);
  __bf16* qk_h    = ah((size_t)T_*DI_);
  __bf16* hf_h    = ah((size_t)T_*D_);
  __bf16* wskip_h = ah((size_t)DM_*DI_);
  __bf16* wup_h   = ah((size_t)DM_*DI_);
  __bf16* convk_h = ah((size_t)4*DI_*DI_);
  __bf16* wi_h    = ah((size_t)64*D_);
  __bf16* wf_h    = ah((size_t)64*D_);
  __bf16* wo_h    = ah((size_t)DI_*D_);
  __bf16* wq_h    = ah((size_t)DI_*D_);
  __bf16* wk_h    = ah((size_t)DI_*D_);
  __bf16* wv_h    = ah((size_t)DI_*D_);
  __bf16* wd_h    = ah((size_t)D_*DM_);

  auto cvt = [&](const float* s, __bf16* d, int n){
    cvt_f32_bf16<<<(n + 255)/256, 256, 0, stream>>>(s, d, n);
  };
  cvt(w_skip2, wskip_h, DM_*DI_);
  cvt(w_up,    wup_h,   DM_*DI_);
  cvt(conv_k,  convk_h, 4*DI_*DI_);
  cvt(wi,      wi_h,    64*D_);
  cvt(wf,      wf_h,    64*D_);
  cvt(wo,      wo_h,    DI_*D_);
  cvt(wq,      wq_h,    DI_*D_);
  cvt(wk,      wk_h,    DI_*D_);
  cvt(wv,      wv_h,    DI_*D_);
  cvt(w_down,  wd_h,    D_*DM_);

  ln_kernel<<<T_, 128, 0, stream>>>(x, ln_s, ln_b, xn_h);

  dim3 wb(32, 8);
  // skip2 = silu(xn @ w_skip2 + b)
  gemm_bf16_kernel<<<128, wb, 0, stream>>>(xn_h, DM_, wskip_h, DI_, b_skip2, DI_-1, 1.f, 1,
                                           nullptr, nullptr, skip2, nullptr, T_, DI_, DM_, 4, 0);
  // xu = xn @ w_up + b
  gemm_bf16_kernel<<<128, wb, 0, stream>>>(xn_h, DM_, wup_h, DI_, b_up, DI_-1, 1.f, 0,
                                           nullptr, nullptr, xu, xu_h, T_, DI_, DM_, 4, 0);
  // qk = silu(conv1d(xu))
  conv_kernel<<<128, wb, 0, stream>>>(xu_h, convk_h, conv_b, qk, qk_h);
  // it = exp(blockdiag(xu @ wi) + bi)
  gemm_bf16_kernel<<<128, wb, 0, stream>>>(xu_h, DI_, wi_h, D_, bi, HD_-1, 1.f, 3,
                                           nullptr, nullptr, itb, nullptr, T_, D_, 64, 4, 1);
  // zf = blockdiag(xu @ wf) + bf   (log-space for stable cumprod)
  gemm_bf16_kernel<<<128, wb, 0, stream>>>(xu_h, DI_, wf_h, D_, bff, HD_-1, 1.f, 0,
                                           nullptr, nullptr, zf, nullptr, T_, D_, 64, 4, 1);
  // ot = sigmoid(xu @ wo + bo)
  gemm_bf16_kernel<<<128, wb, 0, stream>>>(xu_h, DI_, wo_h, D_, bo, D_-1, 1.f, 2,
                                           nullptr, nullptr, otb, nullptr, T_, D_, DI_, 4, 0);
  // q = qk @ wq + bq
  gemm_bf16_kernel<<<128, wb, 0, stream>>>(qk_h, DI_, wq_h, D_, bq, D_-1, 1.f, 0,
                                           nullptr, nullptr, qb, nullptr, T_, D_, DI_, 4, 0);
  // k = (qk @ wk + bk) / D
  gemm_bf16_kernel<<<128, wb, 0, stream>>>(qk_h, DI_, wk_h, D_, bk, D_-1, 1.f/(float)D_, 0,
                                           nullptr, nullptr, kb, nullptr, T_, D_, DI_, 4, 0);
  // v = (xu @ wv + bv) * it
  gemm_bf16_kernel<<<128, wb, 0, stream>>>(xu_h, DI_, wv_h, D_, bv, D_-1, 1.f, 0,
                                           itb, nullptr, vb, nullptr, T_, D_, DI_, 4, 0);

  // chunked mLSTM scan
  s1_kernel<<<NC_, wb, 0, stream>>>(vb, kb, itb, zf, Cl, aE, nl);
  s2_kernel<<<D_, 256, 0, stream>>>(Cl, aE, nl, CTh, npre, outC, outN);
  s3_kernel<<<NC_, wb, 0, stream>>>(qb, kb, vb, itb, zf, otb, CTh, npre, aAll, htb);

  // groupnorm + gate + down projection with residual
  gn_stats_kernel<<<H_, 256, 0, stream>>>(htb, gst);
  build_h_kernel<<<T_, 256, 0, stream>>>(htb, qk, skip2, gn_s, gn_b, gst, hf_h);
  gemm_bf16_kernel<<<64, wb, 0, stream>>>(hf_h, D_, wd_h, DM_, b_down, DM_-1, 1.f, 0,
                                          nullptr, x, outp, nullptr, T_, DM_, D_, 3, 0);
}
